// Model_14121852470205
// MI455X (gfx1250) — matrix operand order repaired and checked by reading, not compiled, not run on hardware
//
#include <hip/hip_runtime.h>

// ---------------- problem constants (from reference setup) ----------------
static constexpr int  BS   = 128;
static constexpr int  SEQ  = 8192;
static constexpr int  DIM  = 32;
static constexpr int  NC   = 3;
static constexpr long long ROWS = (long long)SEQ * BS;   // 1,048,576 GEMM rows
static constexpr int  TILES_PER_WAVE = 8;

#define DT_STEP  4.1666666666666665e-05f   // 1/24000
#define DTLIF_B  0.9f

typedef _Float16 half16 __attribute__((ext_vector_type(16)));
typedef float    float8 __attribute__((ext_vector_type(8)));

__device__ __forceinline__ float heav(float x) { return x > 0.0f ? 1.0f : 0.0f; }

// ---------------------------------------------------------------------------
// Phase 1: RAF + DTLIF recurrence. One wave32 per batch row, lane = dim.
// Writes rspk_h[t][b][d], ru_h[t][b][d] (row-major => GEMM A matrix),
// and dt_spk[t*BS+b] into scratch (the spk_h region, consumed by phase 2).
// x is uniform per wave -> read as float4 scalar loads, 4 steps per load.
// ---------------------------------------------------------------------------
__global__ void raf_dtlif_kernel(const float* __restrict__ x,       // [BS][SEQ]
                                 const float* __restrict__ omegas,  // [2][DIM]
                                 const float* __restrict__ bsp,     // [2][DIM]
                                 const float* __restrict__ thrv,    // [DIM]
                                 float* __restrict__ rspk_out,      // [SEQ][BS][DIM]
                                 float* __restrict__ ru_out,        // [SEQ][BS][DIM]
                                 float* __restrict__ dtspk_ws)      // [SEQ*BS]
{
    const int wave = threadIdx.x >> 5;
    const int lane = threadIdx.x & 31;
    const int b    = blockIdx.x * (blockDim.x >> 5) + wave;

    const float om1 = omegas[lane], om2 = omegas[DIM + lane];
    const float bb1 = bsp[lane],    bb2 = bsp[DIM + lane];
    const float thr = thrv[lane];

    float u = 0.0f, v = 0.0f, spk = 0.0f, dtm = 0.0f;
    bool  t1 = true;

    const float*  xb  = x + (long long)b * SEQ;
    const float4* xb4 = (const float4*)xb;

    for (int t4 = 0; t4 < SEQ / 4; ++t4) {
        if ((t4 & 15) == 0 && (t4 * 4 + 256) < SEQ)
            __builtin_prefetch(xb + t4 * 4 + 256, 0, 1);   // global_prefetch_b8

        const float4 xv = xb4[t4];                         // uniform -> s_load_b128
#pragma unroll
        for (int j = 0; j < 4; ++j) {
            const float xt = (j == 0) ? xv.x : (j == 1) ? xv.y : (j == 2) ? xv.z : xv.w;
            const float curr = fminf(fmaxf(xt, -1.0f), 1.0f);

            // --- dual-regime resonate-and-fire (Euler) ---
            const float om = t1 ? om1 : om2;
            const float bb = t1 ? bb1 : bb2;
            const float ur = u - spk * thr;
            const float un = ur + DT_STEP * (bb * ur - om * v) + curr;
            const float vn = v  + DT_STEP * (om * ur + bb * v);
            const float s  = heav(un - thr);
            t1 = t1 ^ (s > 0.0f);
            u = un; v = vn; spk = s;

            // --- dual-threshold LIF on raw input (same value in every lane) ---
            dtm = DTLIF_B * dtm + xt;
            const float pos = heav(dtm - 1.0f);
            const float neg = heav(-1.0f - dtm);
            dtm = dtm - pos + neg;

            const long long base = (long long)(t4 * 4 + j) * BS + b;
            rspk_out[base * DIM + lane] = s;
            ru_out  [base * DIM + lane] = un;
            if (lane == 0) dtspk_ws[base] = pos - neg;
        }
    }
}

// ---------------------------------------------------------------------------
// Phase 2: readout GEMM via WMMA, 8 tiles (16 rows each) per wave.
//   out1[r][c] = sum_k rspk[r][k]*W[c][k]  +  dt_spk[r]*W[c][32] + bias[c]
// A (16x32 f16) per ISA dense 16-bit A table:
//   lane l: M = l&15, group g = l>>4; half h: K = (h<8 ? 8g+h : 16+8g+(h-8))
// B (32x16 f16), same K order as A:
//   lane n: N = n&15, g = n>>4, half h: K = (h<8 ? 8g+h : 16+8g+(h-8))   (cols >= NC zeroed)
// C/D (16x16 f32): VGPR r, lane n: M = r + 8*(n>>4), N = n&15.
// B / bias / wlast are tile-invariant -> hoisted; invalid lanes get
// bc=wlast=0 so C needs no per-tile selects (no exec divergence pre-WMMA).
// ---------------------------------------------------------------------------
__global__ void readout_gemm_kernel(const float* __restrict__ rspk,   // [ROWS][DIM]
                                    const float* __restrict__ W,      // [NC][DIM+1]
                                    const float* __restrict__ bias,   // [NC]
                                    const float* __restrict__ dtspk,  // [ROWS]
                                    float*       out1)                // [ROWS][NC]
{
    const int wid  = (int)((blockIdx.x * (unsigned)blockDim.x + threadIdx.x) >> 5);
    const int lane = threadIdx.x & 31;
    const int m    = lane & 15;       // A row (M) / B column (N)
    const int g    = lane >> 4;       // lane-half select
    const bool valid = (m < NC);
    const int brow   = valid ? m : 0; // clamped -> unconditional loads

    // ---- B: W[:, :32] column N = m ----
    half16 Bm;
#pragma unroll
    for (int h = 0; h < 16; ++h) {
        const float w = W[brow * (DIM + 1) + (g * 8 + h + (h >= 8 ? 8 : 0))];
        Bm[h] = (_Float16)(valid ? w : 0.0f);
    }
    const float wlast = valid ? W[brow * (DIM + 1) + DIM] : 0.0f;
    const float bc    = valid ? bias[brow] : 0.0f;

    const long long tile0 = (long long)wid * TILES_PER_WAVE;
    for (int tt = 0; tt < TILES_PER_WAVE; ++tt) {
        const long long r0 = (tile0 + tt) * 16;

        // ---- A: 16 spike values (exact in f16) for this lane's (M, K) slots ----
        const float* arow = rspk + (r0 + m) * DIM;
        const float4 a0 = *(const float4*)(arow + g * 8);
        const float4 a1 = *(const float4*)(arow + g * 8 + 4);
        const float4 a2 = *(const float4*)(arow + 16 + g * 8);
        const float4 a3 = *(const float4*)(arow + 16 + g * 8 + 4);
        half16 A;
        A[0]=(_Float16)a0.x;  A[1]=(_Float16)a0.y;  A[2]=(_Float16)a0.z;  A[3]=(_Float16)a0.w;
        A[4]=(_Float16)a1.x;  A[5]=(_Float16)a1.y;  A[6]=(_Float16)a1.z;  A[7]=(_Float16)a1.w;
        A[8]=(_Float16)a2.x;  A[9]=(_Float16)a2.y;  A[10]=(_Float16)a2.z; A[11]=(_Float16)a2.w;
        A[12]=(_Float16)a3.x; A[13]=(_Float16)a3.y; A[14]=(_Float16)a3.z; A[15]=(_Float16)a3.w;

        // ---- C: bias + dt_spk * W[:,32]; rows M = 8g..8g+7 are contiguous ----
        const float4 d0 = *(const float4*)(dtspk + r0 + g * 8);
        const float4 d1 = *(const float4*)(dtspk + r0 + g * 8 + 4);
        float8 C;
        C[0] = fmaf(d0.x, wlast, bc); C[1] = fmaf(d0.y, wlast, bc);
        C[2] = fmaf(d0.z, wlast, bc); C[3] = fmaf(d0.w, wlast, bc);
        C[4] = fmaf(d1.x, wlast, bc); C[5] = fmaf(d1.y, wlast, bc);
        C[6] = fmaf(d1.z, wlast, bc); C[7] = fmaf(d1.w, wlast, bc);

        // ---- D = A x B + C  (EXEC all-ones here) ----
        const float8 D = __builtin_amdgcn_wmma_f32_16x16x32_f16(
            /*neg_a=*/false, A, /*neg_b=*/false, Bm,
            /*c_mod=*/(short)0, C, /*reuse_a=*/false, /*reuse_b=*/false);

        if (valid) {
#pragma unroll
            for (int r = 0; r < 8; ++r)
                out1[(r0 + g * 8 + r) * NC + m] = D[r];
        }
    }
}

// ---------------------------------------------------------------------------
// Phase 3: output leaky LIF scan. One thread per (b, c); 384 threads.
// out1 lives in the mem_h region: each thread reads out1[idx] then overwrites
// the same element with mem -- same-thread read-then-write, so no __restrict__.
// ---------------------------------------------------------------------------
__global__ void lif_out_kernel(float*       out1_and_mem,            // [SEQ][BS][NC]
                               const float* __restrict__ beta_p,
                               const float* __restrict__ thr_p,
                               float* __restrict__ spk_h)            // [SEQ][BS][NC]
{
    const int tid = blockIdx.x * blockDim.x + threadIdx.x;   // = b*NC + c
    if (tid >= BS * NC) return;
    const float beta = beta_p[0];
    const float thr  = thr_p[0];
    float mem = 0.0f;
    for (int t = 0; t < SEQ; ++t) {
        const long long idx = (long long)t * (BS * NC) + tid;
        const float o = out1_and_mem[idx];
        mem = beta * mem + o;
        const float s = heav(mem - thr);
        mem -= s * thr;
        spk_h[idx]        = s;
        out1_and_mem[idx] = mem;    // becomes mem_h
    }
}

// ---------------------------------------------------------------------------
extern "C" void kernel_launch(void* const* d_in, const int* in_sizes, int n_in,
                              void* d_out, int out_size, void* d_ws, size_t ws_size,
                              hipStream_t stream) {
    const float* x        = (const float*)d_in[0];   // (128, 8192)
    const float* omegas   = (const float*)d_in[1];   // (2, 32)
    const float* bsp      = (const float*)d_in[2];   // (2, 32)
    const float* thrv     = (const float*)d_in[3];   // (32,)
    const float* W        = (const float*)d_in[4];   // (3, 33)
    const float* bias     = (const float*)d_in[5];   // (3,)
    const float* lif_beta = (const float*)d_in[6];   // scalar
    const float* lif_thr  = (const float*)d_in[7];   // scalar

    float* out = (float*)d_out;
    const long long n_sm = (long long)SEQ * BS * NC;   // 3,145,728
    float* spk_h  = out;                  // [SEQ][BS][NC]
    float* mem_h  = out + n_sm;           // [SEQ][BS][NC]  (holds out1 between P2/P3)
    float* rspk_h = out + 2 * n_sm;       // [SEQ][BS][DIM]
    float* ru_h   = rspk_h + ROWS * DIM;  // [SEQ][BS][DIM]

    // dt_spk scratch (ROWS floats) fits inside the spk_h region (3*ROWS floats);
    // fully consumed by phase 2 before phase 3 overwrites spk_h.
    float* dt_ws = spk_h;
    float* out1  = mem_h;

    // Phase 1: 128 waves (4 waves / block, 32 blocks), lane = dim.
    hipLaunchKernelGGL(raf_dtlif_kernel, dim3(BS / 4), dim3(128), 0, stream,
                       x, omegas, bsp, thrv, rspk_h, ru_h, dt_ws);

    // Phase 2: 65536 tiles, 8 tiles/wave, 8 waves/block -> 1024 blocks of 256.
    hipLaunchKernelGGL(readout_gemm_kernel,
                       dim3((int)(ROWS / 16 / TILES_PER_WAVE / 8)), dim3(256), 0, stream,
                       rspk_h, W, bias, dt_ws, out1);

    // Phase 3: 384 scans.
    hipLaunchKernelGGL(lif_out_kernel, dim3(3), dim3(128), 0, stream,
                       out1, lif_beta, lif_thr, spk_h);
}
